// DeformableTransformerLayer_61735859912940
// MI455X (gfx1250) — matrix operand order repaired and checked by reading, not compiled, not run on hardware
//
#include <hip/hip_runtime.h>
#include <hip/hip_bf16.h>
#include <math.h>

// ---------------------------------------------------------------------------
// Problem constants (fixed by the reference setup)
// ---------------------------------------------------------------------------
#define BS      8
#define NQ      5376
#define NV      5376
#define DMODEL  256
#define HEADS   8
#define HEAD_DIM 32
#define LEVELS  3
#define POINTS  4
#define MROWS   (BS * NQ)        // 43008
#define KDIM    256              // GEMM K for all projections
#define KT_CNT  (KDIM / 32)      // 8 k-tiles of 32
#define NCAT    (HEADS * LEVELS * POINTS * 2 + HEADS * LEVELS * POINTS) // 192+96=288
#define LN_EPS  1e-5f

typedef __attribute__((ext_vector_type(16))) _Float16 v16h;
typedef __attribute__((ext_vector_type(8)))  _Float16 v8h;
typedef __attribute__((ext_vector_type(8)))  float    v8f;

// ---------------------------------------------------------------------------
// Prep: q16 = f16(query + query_pos), v16 = f16(value)
// ---------------------------------------------------------------------------
__global__ void prep_cast_kernel(const float* __restrict__ query,
                                 const float* __restrict__ qpos,
                                 const float* __restrict__ value,
                                 _Float16* __restrict__ q16,
                                 _Float16* __restrict__ v16) {
    size_t i = (size_t)blockIdx.x * blockDim.x + threadIdx.x;
    q16[i] = (_Float16)(query[i] + qpos[i]);
    v16[i] = (_Float16)value[i];
}

// ---------------------------------------------------------------------------
// Swizzle fp32 weights [K x Nsrc] (K=256) into WMMA-B fragment-native f16:
// dst[ ((nt*KT + kt)*32 + lane)*16 + j ] = W[kt*32 + 8*(lane>>4) + (j&7) + 16*(j>>3)][nt*16 + (lane&15)]
// ---------------------------------------------------------------------------
__global__ void swizzle_w_kernel(const float* __restrict__ W, int Nsrc,
                                 _Float16* __restrict__ dst, int nt_base, int total) {
    int t = blockIdx.x * blockDim.x + threadIdx.x;
    if (t >= total) return;
    int nt_local = t >> 12;           // / (K*16) = /4096
    int rem      = t & 4095;
    int kt       = rem >> 9;          // / 512
    int L        = (rem >> 4) & 31;
    int j        = rem & 15;
    int k        = kt * 32 + ((L >> 4) << 3) + (j & 7) + ((j >> 3) << 4);
    int n        = (nt_local << 4) + (L & 15);
    dst[(((size_t)(nt_base + nt_local) * KT_CNT + kt) * 32 + L) * 16 + j] =
        (_Float16)W[(size_t)k * Nsrc + n];
}

__global__ void cat_bias_kernel(const float* __restrict__ b_off,
                                const float* __restrict__ b_attn,
                                float* __restrict__ bcat) {
    int t = threadIdx.x;
    if (t < 192)      bcat[t] = b_off[t];
    else if (t < 288) bcat[t] = b_attn[t - 192];
}

// ---------------------------------------------------------------------------
// WMMA GEMM: C[M x N] = A_f16[M x 256] * Bsw_f16 + bias.
// Each wave computes a 16 x (16*NTW) tile: one A fragment feeds NTW WMMAs per
// K-step, amortizing activation loads (A is L2-resident; this halves+ L2
// traffic per FLOP vs 1 n-tile/wave). 8 waves per block along M.
// ---------------------------------------------------------------------------
template <int NTW>
__global__ void gemm_wmma_kernel(const _Float16* __restrict__ A,
                                 const _Float16* __restrict__ Bsw,
                                 const float* __restrict__ bias,
                                 float* __restrict__ C, int M, int N) {
    const int lane  = threadIdx.x & 31;
    const int wave  = threadIdx.x >> 5;
    const int mt    = blockIdx.x * 8 + wave;
    const int ntg   = blockIdx.y;            // group of NTW n-tiles
    if (mt * 16 >= M) return;
    const int m     = mt * 16 + (lane & 15);
    const int khalf = lane >> 4;
    const _Float16* arow = A + (size_t)m * KDIM;

    const _Float16* bfrag[NTW];
#pragma unroll
    for (int i = 0; i < NTW; ++i) {
        const int nt = ntg * NTW + i;
        bfrag[i] = Bsw + (((size_t)nt * KT_CNT) * 32 + lane) * 16;
    }

    v8f c[NTW];
#pragma unroll
    for (int i = 0; i < NTW; ++i) c[i] = (v8f){};

#pragma unroll
    for (int kt = 0; kt < KT_CNT; ++kt) {
        const int k0 = kt * 32;
        union { v16h v; v8h h[2]; } a;
        a.h[0] = *(const v8h*)(arow + k0 + 8 * khalf);
        a.h[1] = *(const v8h*)(arow + k0 + 16 + 8 * khalf);
#pragma unroll
        for (int i = 0; i < NTW; ++i) {
            v16h b = *(const v16h*)(bfrag[i] + (size_t)kt * 512);
            c[i] = __builtin_amdgcn_wmma_f32_16x16x32_f16(false, a.v, false, b,
                                                          (short)0, c[i], false, false);
        }
    }

#pragma unroll
    for (int i = 0; i < NTW; ++i) {
        const int n  = (ntg * NTW + i) * 16 + (lane & 15);
        const float bv = bias[n];
#pragma unroll
        for (int r = 0; r < 8; ++r) {
            int row = mt * 16 + r + 8 * khalf;
            C[(size_t)row * N + n] = c[i][r] + bv;
        }
    }
}

// ---------------------------------------------------------------------------
// Deformable sampling: one block per (b,q), thread = (head, dim).
// proj row layout: [0:192) offsets (h,l,p,2), [192:288) attn logits (h, l*4+p)
// ---------------------------------------------------------------------------
__global__ void msda_sample_kernel(const float* __restrict__ proj,
                                   const float* __restrict__ ref_pts,
                                   const float* __restrict__ vproj,
                                   const int* __restrict__ shapes,
                                   const int* __restrict__ lstart,
                                   _Float16* __restrict__ msda) {
    const int bq = blockIdx.x;            // b*NQ + q
    const int b  = bq / NQ;
    const int h  = threadIdx.x >> 5;
    const int d  = threadIdx.x & 31;

    const float* prow = proj + (size_t)bq * NCAT;
    // softmax over the 12 (level,point) logits of this head
    float wts[LEVELS * POINTS];
    float mx = -1e30f;
#pragma unroll
    for (int i = 0; i < LEVELS * POINTS; ++i) {
        wts[i] = prow[192 + h * (LEVELS * POINTS) + i];
        mx = fmaxf(mx, wts[i]);
    }
    float sum = 0.f;
#pragma unroll
    for (int i = 0; i < LEVELS * POINTS; ++i) { wts[i] = expf(wts[i] - mx); sum += wts[i]; }
    const float inv = 1.f / sum;

    const float* rp = ref_pts + (size_t)bq * (LEVELS * 2);
    float acc = 0.f;
#pragma unroll
    for (int l = 0; l < LEVELS; ++l) {
        const int H  = shapes[2 * l];
        const int W  = shapes[2 * l + 1];
        const int st = lstart[l];
        const float rx = rp[2 * l + 0];
        const float ry = rp[2 * l + 1];
        const float* vbase = vproj + ((size_t)(b * NV + st)) * DMODEL + h * HEAD_DIM + d;
#pragma unroll
        for (int p = 0; p < POINTS; ++p) {
            const float ox = prow[h * 24 + l * 8 + p * 2 + 0];
            const float oy = prow[h * 24 + l * 8 + p * 2 + 1];
            const float x = (rx + ox / (float)W) * (float)W - 0.5f;
            const float y = (ry + oy / (float)H) * (float)H - 0.5f;
            const float x0 = floorf(x), y0 = floorf(y);
            const float aw = wts[l * POINTS + p] * inv;
#pragma unroll
            for (int dy = 0; dy <= 1; ++dy)
#pragma unroll
                for (int dx = 0; dx <= 1; ++dx) {
                    const float xi = x0 + dx, yi = y0 + dy;
                    if (xi >= 0.f && xi < (float)W && yi >= 0.f && yi < (float)H) {
                        const float bw = (1.f - fabsf(x - xi)) * (1.f - fabsf(y - yi));
                        const int idx = (int)yi * W + (int)xi;
                        acc += aw * bw * vbase[(size_t)idx * DMODEL];
                    }
                }
        }
    }
    msda[(size_t)bq * DMODEL + h * HEAD_DIM + d] = (_Float16)acc;
}

// ---------------------------------------------------------------------------
// Final: GEMM (msda @ W_out + b_out) + 2*query residual + LayerNorm.
// 512 threads = 16 waves; block owns one 16-row tile, waves span all 16
// n-tiles (N=256). LN done wave-per-row with wave32 shuffles.
// ---------------------------------------------------------------------------
__global__ void gemm_out_ln_kernel(const _Float16* __restrict__ A,
                                   const _Float16* __restrict__ Bsw,
                                   const float* __restrict__ b_out,
                                   const float* __restrict__ query,
                                   const float* __restrict__ ln_g,
                                   const float* __restrict__ ln_b,
                                   float* __restrict__ out) {
    __shared__ float tile[16][DMODEL + 1];
    const int lane  = threadIdx.x & 31;
    const int wave  = threadIdx.x >> 5;
    const int mt    = blockIdx.x;
    const int nt    = wave;                 // 16 waves -> 16 n-tiles
    const int m     = mt * 16 + (lane & 15);
    const int khalf = lane >> 4;

    const _Float16* arow  = A + (size_t)m * KDIM;
    const _Float16* bfrag = Bsw + (((size_t)nt * KT_CNT) * 32 + lane) * 16;

    v8f c = {};
#pragma unroll
    for (int kt = 0; kt < KT_CNT; ++kt) {
        const int k0 = kt * 32;
        union { v16h v; v8h h[2]; } a;
        a.h[0] = *(const v8h*)(arow + k0 + 8 * khalf);
        a.h[1] = *(const v8h*)(arow + k0 + 16 + 8 * khalf);
        v16h b = *(const v16h*)(bfrag + (size_t)kt * 512);
        c = __builtin_amdgcn_wmma_f32_16x16x32_f16(false, a.v, false, b,
                                                   (short)0, c, false, false);
    }

    const int n  = nt * 16 + (lane & 15);
    const float bv = b_out[n];
#pragma unroll
    for (int r = 0; r < 8; ++r) {
        const int rl  = r + 8 * khalf;
        const int row = mt * 16 + rl;
        tile[rl][n] = c[r] + bv + 2.f * query[(size_t)row * DMODEL + n];
    }
    __syncthreads();

    // wave w normalizes row w (256 elements, 8 per lane)
    const int rl  = wave;
    const int row = mt * 16 + rl;
    float s = 0.f, s2 = 0.f;
#pragma unroll
    for (int e = 0; e < 8; ++e) {
        const float x = tile[rl][lane + 32 * e];
        s += x; s2 += x * x;
    }
#pragma unroll
    for (int msk = 16; msk; msk >>= 1) {
        s  += __shfl_xor(s,  msk, 32);
        s2 += __shfl_xor(s2, msk, 32);
    }
    const float mu  = s * (1.f / DMODEL);
    const float var = s2 * (1.f / DMODEL) - mu * mu;
    const float invs = rsqrtf(var + LN_EPS);
#pragma unroll
    for (int e = 0; e < 8; ++e) {
        const int nn = lane + 32 * e;
        const float x = tile[rl][nn];
        out[(size_t)row * DMODEL + nn] = (x - mu) * invs * ln_g[nn] + ln_b[nn];
    }
}

// ---------------------------------------------------------------------------
// Host launcher
// ---------------------------------------------------------------------------
extern "C" void kernel_launch(void* const* d_in, const int* in_sizes, int n_in,
                              void* d_out, int out_size, void* d_ws, size_t ws_size,
                              hipStream_t stream) {
    (void)in_sizes; (void)n_in; (void)out_size; (void)ws_size;
    const float* query  = (const float*)d_in[0];
    const float* value  = (const float*)d_in[1];
    const float* qpos   = (const float*)d_in[2];
    const float* refp   = (const float*)d_in[3];
    const int*   shapes = (const int*)d_in[4];
    const int*   lstart = (const int*)d_in[5];
    const float* W_off  = (const float*)d_in[6];
    const float* b_off  = (const float*)d_in[7];
    const float* W_attn = (const float*)d_in[8];
    const float* b_attn = (const float*)d_in[9];
    const float* W_val  = (const float*)d_in[10];
    const float* b_val  = (const float*)d_in[11];
    const float* W_out  = (const float*)d_in[12];
    const float* b_out  = (const float*)d_in[13];
    const float* ln_g   = (const float*)d_in[14];
    const float* ln_b   = (const float*)d_in[15];
    float* out = (float*)d_out;

    // workspace carve-out (256B aligned)
    char* p = (char*)d_ws;
    auto take = [&](size_t bytes) -> void* {
        void* r = (void*)p;
        p += (bytes + 255) & ~(size_t)255;
        return r;
    };
    _Float16* q16     = (_Float16*)take((size_t)MROWS * DMODEL * 2);
    _Float16* v16     = (_Float16*)take((size_t)MROWS * DMODEL * 2);
    _Float16* Wval_sw = (_Float16*)take((size_t)KDIM * DMODEL * 2);
    _Float16* Wcat_sw = (_Float16*)take((size_t)KDIM * NCAT * 2);
    _Float16* Wout_sw = (_Float16*)take((size_t)KDIM * DMODEL * 2);
    float*    bcat    = (float*)take(NCAT * 4);
    float*    vproj   = (float*)take((size_t)MROWS * DMODEL * 4);
    float*    proj    = (float*)take((size_t)MROWS * NCAT * 4);
    _Float16* msda    = (_Float16*)take((size_t)MROWS * DMODEL * 2);

    // 1) cast/prep activations
    prep_cast_kernel<<<MROWS, 256, 0, stream>>>(query, qpos, value, q16, v16);

    // 2) weight swizzles (frag-native B layout) + bias concat
    swizzle_w_kernel<<<(KDIM * DMODEL + 255) / 256, 256, 0, stream>>>(
        W_val, DMODEL, Wval_sw, 0, KDIM * DMODEL);
    swizzle_w_kernel<<<(KDIM * 192 + 255) / 256, 256, 0, stream>>>(
        W_off, 192, Wcat_sw, 0, KDIM * 192);
    swizzle_w_kernel<<<(KDIM * 96 + 255) / 256, 256, 0, stream>>>(
        W_attn, 96, Wcat_sw, 12, KDIM * 96);
    swizzle_w_kernel<<<(KDIM * DMODEL + 255) / 256, 256, 0, stream>>>(
        W_out, DMODEL, Wout_sw, 0, KDIM * DMODEL);
    cat_bias_kernel<<<1, 288, 0, stream>>>(b_off, b_attn, bcat);

    // 3) value projection GEMM: (43008 x 256) @ (256 x 256), 16x64 per wave
    gemm_wmma_kernel<4><<<dim3(MROWS / 128, DMODEL / 64), 256, 0, stream>>>(
        v16, Wval_sw, b_val, vproj, MROWS, DMODEL);

    // 4) fused offsets+attn GEMM: (43008 x 256) @ (256 x 288), 16x48 per wave
    gemm_wmma_kernel<3><<<dim3(MROWS / 128, NCAT / 48), 256, 0, stream>>>(
        q16, Wcat_sw, bcat, proj, MROWS, NCAT);

    // 5) deformable bilinear sampling + softmax-weighted aggregation
    msda_sample_kernel<<<MROWS, 256, 0, stream>>>(proj, refp, vproj, shapes, lstart, msda);

    // 6) output GEMM + residual (2*query) + LayerNorm, fused
    gemm_out_ln_kernel<<<MROWS / 16, 512, 0, stream>>>(
        msda, Wout_sw, b_out, query, ln_g, ln_b, out);
}
